// SageConv_30812095381967
// MI455X (gfx1250) — matrix operand order repaired and checked by reading, not compiled, not run on hardware
//
#include <hip/hip_runtime.h>
#include <hip/hip_bf16.h>

typedef __attribute__((ext_vector_type(16))) __bf16 v16bf;
typedef __attribute__((ext_vector_type(8)))  __bf16 v8bf;
typedef __attribute__((ext_vector_type(8)))  float  v8f;
typedef __attribute__((ext_vector_type(4)))  float  v4f;

#define N_NODES 16384
#define D_FEAT  256
#define D_OUT   256
#define K2      512   /* 2*D */

// workspace layout (bytes)
#define FEATB_TILES 8192ull                 /* 512 ktiles x 16 ntiles */
#define WB_TILES    256ull                  /* 16 ktiles x 16 ntiles  */
#define FEATB_OFF   0ull
#define FEATB_BYTES (FEATB_TILES * 512ull * 2ull)   /* 8 MiB   */
#define WB_OFF      (FEATB_OFF + FEATB_BYTES)
#define WB_BYTES    (WB_TILES * 512ull * 2ull)      /* 256 KiB */
#define NEIGH_OFF   (WB_OFF + WB_BYTES)             /* 8 MiB bf16 [N, D] */

static __device__ __forceinline__ __bf16 tobf(float x) { return (__bf16)x; }

// ---------------------------------------------------------------------------
// Prepack features and W^T into WMMA B-fragment tile layout (bf16).
// Tile (kt, nt) covers K = kt*32..+31, N = nt*16..+15. Storage: 32 lanes x
// 16 bf16 (32B) contiguous per lane, so a GEMM lane loads its whole fragment
// with one 32-byte (2 x b128) load.
//   B-layout (16-bit, 32x16): lane = ((K>>3)&1)*16 + N;  element p = (K & 7) + 8*(K>=16).
// ---------------------------------------------------------------------------
__global__ void __launch_bounds__(256)
prepack_kernel(const float* __restrict__ feat, const float* __restrict__ W,
               __bf16* __restrict__ featB, __bf16* __restrict__ WB) {
  const int wave = threadIdx.x >> 5;
  const int lane = threadIdx.x & 31;
  const int gtile = blockIdx.x * 8 + wave;
  const int ln = lane & 15;
  const int hi = lane >> 4;
  if (gtile < (int)FEATB_TILES) {
    // B = features (K x N), row-major [K][N]
    const int kt = gtile >> 4, nt = gtile & 15;
    const int n  = nt * 16 + ln;
    const int kb = kt * 32 + hi * 8;
    v16bf v;
#pragma unroll
    for (int p = 0; p < 16; ++p)
      v[p] = tobf(feat[(size_t)(kb + p + (p & 8)) * D_FEAT + n]);
    *(v16bf*)(featB + (size_t)gtile * 512 + lane * 16) = v;
  } else if (gtile < (int)(FEATB_TILES + WB_TILES)) {
    // B = W^T, i.e. B[k][n] = W[n][k], W row-major [OUT][K2]
    const int t  = gtile - (int)FEATB_TILES;
    const int kt = t >> 4, nt = t & 15;
    const int n  = nt * 16 + ln;
    const int kb = kt * 32 + hi * 8;
    v16bf v;
#pragma unroll
    for (int p = 0; p < 16; ++p)
      v[p] = tobf(W[(size_t)n * K2 + kb + p + (p & 8)]);
    *(v16bf*)(WB + (size_t)t * 512 + lane * 16) = v;
  }
}

// ---------------------------------------------------------------------------
// GEMM1: neigh = (adj @ features) / (rowsum(adj) + 1), written as bf16.
// Workgroup = 256 threads = 8 waves (wm 0..3, wn 0..1), tile 64 rows x 256
// cols, K = 16384 in steps of 32. Row-sums accumulate into four v4f vector
// accumulators (pair-aligned -> clean v_pk_add_f32, no register shuffling),
// horizontally reduced after the K loop. adj is a 1 GiB single-use stream
// -> non-temporal loads so featB (8 MiB) stays resident in L2.
//   A-layout (16-bit, 16x32): lane = ksec*16 + M; elements 0..7 hold
//   K = bk..bk+7, elements 8..15 hold K = bk+16..bk+23, bk = k0 + 8*ksec.
// ---------------------------------------------------------------------------
__global__ void __launch_bounds__(256)
gemm1_kernel(const float* __restrict__ adj, const __bf16* __restrict__ featB,
             __bf16* __restrict__ neighb) {
  const int tid  = threadIdx.x;
  const int wave = tid >> 5, lane = tid & 31;
  const int wm = wave >> 1, wn = wave & 1;
  const int ml = lane & 15, hi = lane >> 4;
  const int rowblk = blockIdx.x * 64;
  const int arow   = rowblk + wm * 16 + ml;
  const float* ap  = adj + (size_t)arow * N_NODES;

  v8f acc[8] = {};
  v4f vs0 = {}, vs1 = {}, vs2 = {}, vs3 = {};

  for (int kt = 0; kt < 512; ++kt) {
    const int bk = kt * 32 + 8 * hi;
    const v4f f0 = __builtin_nontemporal_load((const v4f*)(ap + bk));
    const v4f f1 = __builtin_nontemporal_load((const v4f*)(ap + bk + 4));
    const v4f f2 = __builtin_nontemporal_load((const v4f*)(ap + bk + 16));
    const v4f f3 = __builtin_nontemporal_load((const v4f*)(ap + bk + 20));

    vs0 += f0;   // v_pk_add_f32 pairs, operands already pair-aligned
    vs1 += f1;
    vs2 += f2;
    vs3 += f3;

    v16bf a;
    a[0]  = tobf(f0.x); a[1]  = tobf(f0.y); a[2]  = tobf(f0.z); a[3]  = tobf(f0.w);
    a[4]  = tobf(f1.x); a[5]  = tobf(f1.y); a[6]  = tobf(f1.z); a[7]  = tobf(f1.w);
    a[8]  = tobf(f2.x); a[9]  = tobf(f2.y); a[10] = tobf(f2.z); a[11] = tobf(f2.w);
    a[12] = tobf(f3.x); a[13] = tobf(f3.y); a[14] = tobf(f3.z); a[15] = tobf(f3.w);

    const __bf16* bt = featB + ((size_t)kt * 16 + wn * 8) * 512 + lane * 16;
#pragma unroll
    for (int nt = 0; nt < 8; ++nt) {
      v16bf b = *(const v16bf*)(bt + (size_t)nt * 512);
      acc[nt] = __builtin_amdgcn_wmma_f32_16x16x32_bf16(
          false, a, false, b, (short)0, acc[nt], false, false);
    }
  }

  // horizontal reduction of the vector row-sum accumulators
  const v4f vs = (vs0 + vs1) + (vs2 + vs3);
  float rs = (vs.x + vs.y) + (vs.z + vs.w);
  // combine half-K partials: lane L + lane L^16 cover the full row
  rs += __shfl_xor(rs, 16, 32);

  __shared__ float deg_s[64];
  if (wn == 0 && lane < 16) deg_s[wm * 16 + lane] = rs + 1.0f;
  __syncthreads();

  float dinv[8];
#pragma unroll
  for (int r = 0; r < 8; ++r)
    dinv[r] = 1.0f / deg_s[wm * 16 + 8 * hi + r];

  // C-layout: VGPR r, lane -> row = r + 8*hi, col = lane&15
#pragma unroll
  for (int nt = 0; nt < 8; ++nt) {
    const int col = wn * 128 + nt * 16 + ml;
#pragma unroll
    for (int r = 0; r < 8; ++r) {
      const int row = rowblk + wm * 16 + 8 * hi + r;
      neighb[(size_t)row * D_FEAT + col] = tobf(acc[nt][r] * dinv[r]);
    }
  }
}

// ---------------------------------------------------------------------------
// GEMM2: out = [features | neigh] @ W^T   (K = 512, f32 output)
// Same 64x256 tiling. A from features (f32 -> bf16 cvt) for k<256, straight
// bf16 16B loads from neigh for k>=256. B from prepacked WB. Output is
// never re-read -> non-temporal stores.
// ---------------------------------------------------------------------------
__global__ void __launch_bounds__(256)
gemm2_kernel(const float* __restrict__ feat, const __bf16* __restrict__ neighb,
             const __bf16* __restrict__ WB, float* __restrict__ out) {
  const int tid  = threadIdx.x;
  const int wave = tid >> 5, lane = tid & 31;
  const int wm = wave >> 1, wn = wave & 1;
  const int ml = lane & 15, hi = lane >> 4;
  const int rowblk = blockIdx.x * 64;
  const int arow   = rowblk + wm * 16 + ml;

  v8f acc[8] = {};
  union AFrag { v16bf v; v8bf h[2]; };

#pragma unroll
  for (int kt = 0; kt < 16; ++kt) {
    const int bk = kt * 32 + 8 * hi;
    AFrag a;
    if (kt < 8) {  // uniform branch: features half (f32 in memory)
      const float* apf = feat + (size_t)arow * D_FEAT;
      const v4f f0 = *(const v4f*)(apf + bk);
      const v4f f1 = *(const v4f*)(apf + bk + 4);
      const v4f f2 = *(const v4f*)(apf + bk + 16);
      const v4f f3 = *(const v4f*)(apf + bk + 20);
      a.v[0]  = tobf(f0.x); a.v[1]  = tobf(f0.y); a.v[2]  = tobf(f0.z); a.v[3]  = tobf(f0.w);
      a.v[4]  = tobf(f1.x); a.v[5]  = tobf(f1.y); a.v[6]  = tobf(f1.z); a.v[7]  = tobf(f1.w);
      a.v[8]  = tobf(f2.x); a.v[9]  = tobf(f2.y); a.v[10] = tobf(f2.z); a.v[11] = tobf(f2.w);
      a.v[12] = tobf(f3.x); a.v[13] = tobf(f3.y); a.v[14] = tobf(f3.z); a.v[15] = tobf(f3.w);
    } else {       // neigh half (bf16 in memory): elements 0..7 = K bk..bk+7,
                   // elements 8..15 = K bk+16..bk+23 -> two contiguous 16B loads
      const __bf16* np = neighb + (size_t)arow * D_FEAT + (bk - 256);
      a.h[0] = *(const v8bf*)(np);
      a.h[1] = *(const v8bf*)(np + 16);
    }
    const __bf16* bt = WB + ((size_t)kt * 16 + wn * 8) * 512 + lane * 16;
#pragma unroll
    for (int nt = 0; nt < 8; ++nt) {
      v16bf b = *(const v16bf*)(bt + (size_t)nt * 512);
      acc[nt] = __builtin_amdgcn_wmma_f32_16x16x32_bf16(
          false, a.v, false, b, (short)0, acc[nt], false, false);
    }
  }

#pragma unroll
  for (int nt = 0; nt < 8; ++nt) {
    const int col = wn * 128 + nt * 16 + ml;
#pragma unroll
    for (int r = 0; r < 8; ++r) {
      const int row = rowblk + wm * 16 + 8 * hi + r;
      __builtin_nontemporal_store(acc[nt][r], &out[(size_t)row * D_OUT + col]);
    }
  }
}

extern "C" void kernel_launch(void* const* d_in, const int* in_sizes, int n_in,
                              void* d_out, int out_size, void* d_ws, size_t ws_size,
                              hipStream_t stream) {
  const float* feat = (const float*)d_in[0];   // [16384, 256]
  const float* adj  = (const float*)d_in[1];   // [16384, 16384]
  const float* W    = (const float*)d_in[2];   // [256, 512]
  float* out = (float*)d_out;                  // [16384, 256]

  char* ws = (char*)d_ws;
  __bf16* featB  = (__bf16*)(ws + FEATB_OFF);
  __bf16* WB     = (__bf16*)(ws + WB_OFF);
  __bf16* neighb = (__bf16*)(ws + NEIGH_OFF);

  (void)in_sizes; (void)n_in; (void)out_size; (void)ws_size;

  prepack_kernel<<<(FEATB_TILES + WB_TILES) / 8, 256, 0, stream>>>(feat, W, featB, WB);
  gemm1_kernel<<<N_NODES / 64, 256, 0, stream>>>(adj, featB, neighb);
  gemm2_kernel<<<N_NODES / 64, 256, 0, stream>>>(feat, neighb, WB, out);
}
